// MHA_GQA_18433999634691
// MI455X (gfx1250) — matrix operand order repaired and checked by reading, not compiled, not run on hardware
//
#include <hip/hip_runtime.h>
#include <hip/hip_bf16.h>

typedef __attribute__((ext_vector_type(16))) __bf16 v16bf;
typedef __attribute__((ext_vector_type(8)))  float  v8f;

#define T_SEQ  2048
#define DMODEL 5120
#define N_H    40
#define N_KV   10
#define HDIM   128
#define QKV_N  7680   // DMODEL + 2*N_KV*HDIM
#define WIN    1024

union Frag { v16bf v; uint4 q[2]; };

__device__ __forceinline__ unsigned short f2bf(float f) {
  unsigned int u = __float_as_uint(f);
  u += 0x7FFFu + ((u >> 16) & 1u);          // round-to-nearest-even
  return (unsigned short)(u >> 16);
}
__device__ __forceinline__ float bf2f(unsigned short h) {
  return __uint_as_float(((unsigned int)h) << 16);
}
__device__ __forceinline__ v8f vzero() {
  v8f z = {0.f, 0.f, 0.f, 0.f, 0.f, 0.f, 0.f, 0.f};
  return z;
}
__device__ __forceinline__ v8f wmma_bf16(v16bf a, v16bf b, v8f c) {
  return __builtin_amdgcn_wmma_f32_16x16x32_bf16(false, a, false, b, (short)0, c,
                                                 false, false);
}

// ---------------------------------------------------------------- f32 -> bf16
__global__ void cvt_f32_bf16(const float4* __restrict__ in,
                             ushort4* __restrict__ out, int n4) {
  int i = blockIdx.x * 256 + threadIdx.x;
  if (i < n4) {
    float4 f = in[i];
    ushort4 o;
    o.x = f2bf(f.x); o.y = f2bf(f.y); o.z = f2bf(f.z); o.w = f2bf(f.w);
    out[i] = o;
  }
}

// --------------------------------------------------- WMMA "NT" GEMM (bf16*bf16)
// C[M,N] = A[M,K] * W[N,K]^T.  Block tile 64x256 (8 waves, 2x4), wave tile 32x64.
template <bool OUT_F32>
__global__ __launch_bounds__(256) void gemm_bf16_nt(
    const unsigned short* __restrict__ A, const unsigned short* __restrict__ W,
    void* __restrict__ C, int M, int N, int K) {
  const int lane  = threadIdx.x & 31;
  const int wave  = threadIdx.x >> 5;
  const int lrow  = lane & 15;
  const int lhalf = lane >> 4;
  const int mbase = blockIdx.y * 64  + (wave & 1) * 32;
  const int nbase = blockIdx.x * 256 + (wave >> 1) * 64;

  v8f acc[2][4];
#pragma unroll
  for (int i = 0; i < 2; ++i)
#pragma unroll
    for (int j = 0; j < 4; ++j) acc[i][j] = vzero();

  const unsigned short* aRow[2];
  const unsigned short* bRow[4];
#pragma unroll
  for (int i = 0; i < 2; ++i)
    aRow[i] = A + (size_t)(mbase + i * 16 + lrow) * K + lhalf * 8;
#pragma unroll
  for (int j = 0; j < 4; ++j)
    bRow[j] = W + (size_t)(nbase + j * 16 + lrow) * K + lhalf * 8;

  for (int k0 = 0; k0 < K; k0 += 32) {
    Frag a[2], b[4];
#pragma unroll
    for (int i = 0; i < 2; ++i) {
      a[i].q[0] = *(const uint4*)(aRow[i] + k0);
      a[i].q[1] = *(const uint4*)(aRow[i] + k0 + 16);
    }
#pragma unroll
    for (int j = 0; j < 4; ++j) {
      b[j].q[0] = *(const uint4*)(bRow[j] + k0);
      b[j].q[1] = *(const uint4*)(bRow[j] + k0 + 16);
      __builtin_prefetch(bRow[j] + k0 + 64, 0, 1);   // global_prefetch_b8
    }
#pragma unroll
    for (int i = 0; i < 2; ++i)
#pragma unroll
      for (int j = 0; j < 4; ++j)
        acc[i][j] = wmma_bf16(a[i].v, b[j].v, acc[i][j]);
  }

#pragma unroll
  for (int i = 0; i < 2; ++i)
#pragma unroll
    for (int j = 0; j < 4; ++j)
#pragma unroll
      for (int r = 0; r < 8; ++r) {
        int row = mbase + i * 16 + r + 8 * lhalf;
        int col = nbase + j * 16 + lrow;
        float v = acc[i][j][r];
        if (OUT_F32)
          ((float*)C)[(size_t)row * N + col] = v;
        else
          ((unsigned short*)C)[(size_t)row * N + col] = f2bf(v);
      }
}

// ------------------------------------------------------------------ RoPE (Q,K)
__global__ void rope_kernel(unsigned short* __restrict__ qkv) {
  int idx = blockIdx.x * 256 + threadIdx.x;
  const int total = T_SEQ * (N_H + N_KV) * (HDIM / 2);
  if (idx >= total) return;
  int i  = idx & 63;
  int hh = (idx >> 6) % (N_H + N_KV);
  int t  = idx / ((N_H + N_KV) * 64);
  int col = (hh < N_H) ? hh * HDIM + 2 * i
                       : DMODEL + (hh - N_H) * HDIM + 2 * i;
  unsigned short* p = qkv + (size_t)t * QKV_N + col;
  // inv_freq = 250000^(-i/64) ; log2(250000) = 17.931568569324174
  float inv = exp2f(-(float)i * (17.931568569324174f / 64.0f));
  float ang = (float)t * inv;
  float s, c;
  sincosf(ang, &s, &c);
  float a = bf2f(p[0]), b = bf2f(p[1]);
  p[0] = f2bf(a * c - b * s);
  p[1] = f2bf(a * s + b * c);
}

// ------------------------------------------------- V transpose: [kvh][hd][t]
__global__ void vtrans_kernel(const unsigned short* __restrict__ qkv,
                              unsigned short* __restrict__ vt) {
  int idx = blockIdx.x * 256 + threadIdx.x;
  const int total = T_SEQ * N_KV * HDIM;
  if (idx >= total) return;
  int d   = idx & (HDIM - 1);
  int kvh = (idx >> 7) % N_KV;
  int t   = idx / (N_KV * HDIM);
  vt[((size_t)kvh * HDIM + d) * T_SEQ + t] =
      qkv[(size_t)t * QKV_N + DMODEL + N_KV * HDIM + kvh * HDIM + d];
}

// ------------------------------------------------ flash attention (per wave)
// One wave handles one (head, 16-query block). 8 waves / block.
__global__ __launch_bounds__(256) void attn_kernel(
    const unsigned short* __restrict__ qkv, const unsigned short* __restrict__ vt,
    unsigned short* __restrict__ attn) {
  __shared__ unsigned short pshm[8 * 16 * 32];
  const int lane  = threadIdx.x & 31;
  const int wave  = threadIdx.x >> 5;
  const int lrow  = lane & 15;
  const int lhalf = lane >> 4;
  const int job   = blockIdx.x * 8 + wave;  // 0..5119
  const int h     = job >> 7;               // query head
  const int qb    = (job & 127) * 16;       // query block start
  const int kvh   = h >> 2;                 // GQA: 4 Q heads per KV head
  unsigned short* myshm = pshm + wave * (16 * 32);

  // Q fragments: 4 k-steps of 32 over HD=128 (A layout).
  Frag qf[4];
#pragma unroll
  for (int kk = 0; kk < 4; ++kk) {
    const unsigned short* p =
        qkv + (size_t)(qb + lrow) * QKV_N + h * HDIM + kk * 32 + lhalf * 8;
    qf[kk].q[0] = *(const uint4*)p;
    qf[kk].q[1] = *(const uint4*)(p + 16);
  }

  v8f o[8];
  float m[8], l[8];
#pragma unroll
  for (int nt = 0; nt < 8; ++nt) o[nt] = vzero();
#pragma unroll
  for (int r = 0; r < 8; ++r) { m[r] = -1e30f; l[r] = 0.f; }

  int kstart = qb - (WIN - 1);
  if (kstart < 0) kstart = 0;
  kstart &= ~31;
  const float scale = 0.08838834764831845f;  // 1/sqrt(128)

  for (int kb = kstart; kb <= qb + 15; kb += 32) {   // kb <= 2016 -> in bounds
    // ---- S = Q * K^T for 32 keys (two 16x16 C tiles), K-dim = 128
    v8f c0 = vzero(), c1 = vzero();
#pragma unroll
    for (int kk = 0; kk < 4; ++kk) {
      Frag kf;
      const unsigned short* p0 = qkv + (size_t)(kb + lrow) * QKV_N + DMODEL +
                                 kvh * HDIM + kk * 32 + lhalf * 8;
      kf.q[0] = *(const uint4*)p0;
      kf.q[1] = *(const uint4*)(p0 + 16);
      c0 = wmma_bf16(qf[kk].v, kf.v, c0);
      const unsigned short* p1 = p0 + (size_t)16 * QKV_N;
      kf.q[0] = *(const uint4*)p1;
      kf.q[1] = *(const uint4*)(p1 + 16);
      c1 = wmma_bf16(qf[kk].v, kf.v, c1);
    }

    // ---- mask + scale
    float s0[8], s1[8];
    const int key0 = kb + lrow;
    const int key1 = kb + 16 + lrow;
#pragma unroll
    for (int r = 0; r < 8; ++r) {
      const int qrow = qb + r + 8 * lhalf;
      float v0 = c0[r] * scale;
      float v1 = c1[r] * scale;
      if (key0 > qrow || (qrow - key0) >= WIN) v0 = -1e38f;
      if (key1 > qrow || (qrow - key1) >= WIN) v1 = -1e38f;
      s0[r] = v0; s1[r] = v1;
    }

    // ---- online softmax (row lives across a 16-lane group)
#pragma unroll
    for (int r = 0; r < 8; ++r) {
      float rm = fmaxf(s0[r], s1[r]);
      rm = fmaxf(rm, __shfl_xor(rm, 1));
      rm = fmaxf(rm, __shfl_xor(rm, 2));
      rm = fmaxf(rm, __shfl_xor(rm, 4));
      rm = fmaxf(rm, __shfl_xor(rm, 8));
      float mnew = fmaxf(m[r], rm);
      float corr = __expf(m[r] - mnew);
      float p0 = __expf(s0[r] - mnew);
      float p1 = __expf(s1[r] - mnew);
      float rs = p0 + p1;
      rs += __shfl_xor(rs, 1);
      rs += __shfl_xor(rs, 2);
      rs += __shfl_xor(rs, 4);
      rs += __shfl_xor(rs, 8);
      l[r] = l[r] * corr + rs;
      m[r] = mnew;
#pragma unroll
      for (int nt = 0; nt < 8; ++nt) o[nt][r] *= corr;
      s0[r] = p0; s1[r] = p1;
    }

    // ---- C-layout P -> A-layout P via wave-private LDS tile [16][32]
#pragma unroll
    for (int r = 0; r < 8; ++r) {
      int row = r + 8 * lhalf;
      myshm[row * 32 + lrow]      = f2bf(s0[r]);
      myshm[row * 32 + 16 + lrow] = f2bf(s1[r]);
    }
    Frag pf;
    pf.q[0] = *(const uint4*)(myshm + lrow * 32 + lhalf * 8);
    pf.q[1] = *(const uint4*)(myshm + lrow * 32 + lhalf * 8 + 16);

    // ---- O += P * V  (V transposed: contiguous along keys)
#pragma unroll
    for (int nt = 0; nt < 8; ++nt) {
      Frag vf;
      const unsigned short* p =
          vt + ((size_t)kvh * HDIM + nt * 16 + lrow) * T_SEQ + kb + lhalf * 8;
      vf.q[0] = *(const uint4*)p;
      vf.q[1] = *(const uint4*)(p + 16);
      o[nt] = wmma_bf16(pf.v, vf.v, o[nt]);
    }
  }

  // ---- normalize + store bf16 into attn [T, 5120]
#pragma unroll
  for (int nt = 0; nt < 8; ++nt)
#pragma unroll
    for (int r = 0; r < 8; ++r) {
      float val = o[nt][r] / l[r];
      int row = qb + r + 8 * lhalf;
      attn[(size_t)row * DMODEL + h * HDIM + nt * 16 + lrow] = f2bf(val);
    }
}

// ----------------------------------------------------------------- launcher
extern "C" void kernel_launch(void* const* d_in, const int* in_sizes, int n_in,
                              void* d_out, int out_size, void* d_ws,
                              size_t ws_size, hipStream_t stream) {
  (void)in_sizes; (void)n_in; (void)out_size; (void)ws_size;
  const float* x    = (const float*)d_in[0];
  const float* Wqkv = (const float*)d_in[1];
  const float* Wout = (const float*)d_in[2];
  float* out = (float*)d_out;

  unsigned short* xb    = (unsigned short*)d_ws;                 // 2048*5120
  unsigned short* wqkvb = xb    + (size_t)T_SEQ * DMODEL;        // 7680*5120
  unsigned short* woutb = wqkvb + (size_t)QKV_N * DMODEL;        // 5120*5120
  unsigned short* qkvb  = woutb + (size_t)DMODEL * DMODEL;       // 2048*7680
  unsigned short* vtb   = qkvb  + (size_t)T_SEQ * QKV_N;         // 10*128*2048
  unsigned short* attnb = vtb   + (size_t)N_KV * HDIM * T_SEQ;   // 2048*5120

  // 1) fp32 -> bf16 conversions
  {
    int n4 = (T_SEQ * DMODEL) / 4;
    cvt_f32_bf16<<<(n4 + 255) / 256, 256, 0, stream>>>((const float4*)x,
                                                       (ushort4*)xb, n4);
    n4 = (QKV_N * DMODEL) / 4;
    cvt_f32_bf16<<<(n4 + 255) / 256, 256, 0, stream>>>((const float4*)Wqkv,
                                                       (ushort4*)wqkvb, n4);
    n4 = (DMODEL * DMODEL) / 4;
    cvt_f32_bf16<<<(n4 + 255) / 256, 256, 0, stream>>>((const float4*)Wout,
                                                       (ushort4*)woutb, n4);
  }

  // 2) qkv = x @ W_qkv^T   [2048 x 7680]
  gemm_bf16_nt<false><<<dim3(QKV_N / 256, T_SEQ / 64), 256, 0, stream>>>(
      xb, wqkvb, qkvb, T_SEQ, QKV_N, DMODEL);

  // 3) RoPE in place on Q and K sections
  {
    int tot = T_SEQ * (N_H + N_KV) * (HDIM / 2);
    rope_kernel<<<(tot + 255) / 256, 256, 0, stream>>>(qkvb);
  }

  // 4) transpose V -> [kvh][hd][t]
  {
    int tot = T_SEQ * N_KV * HDIM;
    vtrans_kernel<<<(tot + 255) / 256, 256, 0, stream>>>(qkvb, vtb);
  }

  // 5) sliding-window GQA flash attention (40 heads x 128 query blocks)
  attn_kernel<<<(N_H * (T_SEQ / 16)) / 8, 256, 0, stream>>>(qkvb, vtb, attnb);

  // 6) out = attn @ W_out^T   [2048 x 5120], fp32 output
  gemm_bf16_nt<true><<<dim3(DMODEL / 256, T_SEQ / 64), 256, 0, stream>>>(
      attnb, woutb, out, T_SEQ, DMODEL, DMODEL);
}
